// RegionProposalNetwork_24781961298155
// MI455X (gfx1250) — matrix operand order repaired and checked by reading, not compiled, not run on hardware
//
#include <hip/hip_runtime.h>

// ---------------- types ----------------
typedef __attribute__((ext_vector_type(16))) __bf16   v16bf;
typedef __attribute__((ext_vector_type(8)))  float    v8f;
typedef __attribute__((ext_vector_type(4)))  unsigned u32x4;
typedef __attribute__((ext_vector_type(4)))  float    f32x4;
typedef __attribute__((ext_vector_type(4)))  int      v4i;

union UFrag { u32x4 u[2]; v16bf v; };

// ---------------- CDNA5 async global->LDS support (probe-guarded) ----------
#if defined(__has_builtin)
#  if __has_builtin(__builtin_amdgcn_global_load_async_to_lds_b128)
#    define HAVE_ASYNC_LDS 1
#  endif
#endif

#ifdef HAVE_ASYNC_LDS
typedef __attribute__((address_space(1))) v4i as1_v4i;
typedef __attribute__((address_space(3))) v4i as3_v4i;
static __device__ __forceinline__ void async_cp16(const void* g, void* l) {
  __builtin_amdgcn_global_load_async_to_lds_b128(
      (as1_v4i*)g, (as3_v4i*)l, 0, 0);
}
#  if __has_builtin(__builtin_amdgcn_s_wait_asynccnt)
#    define WAIT_ASYNC0() __builtin_amdgcn_s_wait_asynccnt(0)
#  else
#    define WAIT_ASYNC0() asm volatile("s_wait_asynccnt 0" ::: "memory")
#  endif
#else
#  define WAIT_ASYNC0() ((void)0)
#endif

// ---------------- problem dims ----------------
#define HDIM     128
#define WDIM     128
#define S_TOT    16384          // H*W
#define C_INC    1024
#define C_MIDC   512
#define KDIM     9216           // C_IN * 9
#define BKCONV   64             // conv K-step (64 | 1024, never crosses a tap)
#define KT2      144            // KDIM / BKCONV
#define N_ANCH   147456         // S_TOT * 9
#define NPAD     262144         // 2^18 >= N_ANCH
#define PRE_NMSN 6000
#define POST_NMSN 300
#define MASKW    188            // ceil(6000/32)
#define NMS_TH   0.7f

// d_out float offsets: rpn_score(1,147456,2) | rpn_loc(1,147456,4) | rois(1,300,4)
#define SCORE_OFF 0
#define LOC_OFF   294912
#define ROI_OFF   884736

// ---------------- workspace byte offsets ----------------
#define XT_OFF    ((size_t)0)           // 33554432 (reused after conv)
#define BT_OFF    ((size_t)33554432)    //  9437184
#define FEAT_OFF  ((size_t)42991616)    // 16777216
#define WH_OFF    ((size_t)59768832)    //    65536
#define BIAS_OFF  ((size_t)59834368)    //      256
// reuse of XT region after conv:
#define BOX_OFF   ((size_t)0)
#define KEY_OFF   ((size_t)2359296)
#define IDX_OFF   ((size_t)3407872)
#define BS_OFF    ((size_t)4456448)
#define MSK_OFF   ((size_t)4552448)

// ---------------- prep kernels ----------------
__global__ void k_prep_x(const float* __restrict__ x, __bf16* __restrict__ xT) {
  int idx = blockIdx.x * blockDim.x + threadIdx.x;
  if (idx >= S_TOT * C_INC) return;
  int s = idx >> 10;
  int c = idx & 1023;
  xT[idx] = (__bf16)x[(size_t)c * S_TOT + s];
}

__global__ void k_prep_bt(const float* __restrict__ Wc, __bf16* __restrict__ Bt) {
  int idx = blockIdx.x * blockDim.x + threadIdx.x;
  if (idx >= C_MIDC * KDIM) return;
  int n = idx / KDIM;
  int k = idx - n * KDIM;
  int t = k >> 10;
  int c = k & 1023;
  Bt[idx] = (__bf16)Wc[(size_t)n * KDIM + c * 9 + t];
}

__global__ void k_prep_head(const float* __restrict__ Ws, const float* __restrict__ bs,
                            const float* __restrict__ Wl, const float* __restrict__ bl,
                            __bf16* __restrict__ Wh, float* __restrict__ biasAll) {
  int idx = blockIdx.x * blockDim.x + threadIdx.x;
  if (idx >= 64 * 512) return;
  int n = idx >> 9;
  int k = idx & 511;
  float v = 0.f;
  if (n < 18)      v = Ws[n * 512 + k];
  else if (n < 54) v = Wl[(n - 18) * 512 + k];
  Wh[idx] = (__bf16)v;
  if (idx < 64) {
    float b = 0.f;
    if (idx < 18)      b = bs[idx];
    else if (idx < 54) b = bl[idx - 18];
    biasAll[idx] = b;
  }
}

// ---------------- 3x3 conv as implicit GEMM (WMMA bf16) ----------------
// block tile 128(M) x 128(N), 8 waves (2x4), wave tile 64x32,
// K-step 64, double-buffered LDS, async global->LDS prefetch when available.
__global__ __launch_bounds__(256) void k_conv_wmma(
    const __bf16* __restrict__ xT, const __bf16* __restrict__ Bt,
    const float* __restrict__ bconv, __bf16* __restrict__ feat) {
  __shared__ __align__(16) __bf16 As[2][128][BKCONV];
  __shared__ __align__(16) __bf16 Bs[2][128][BKCONV];

  const int tid  = threadIdx.x;
  const int lane = tid & 31;
  const int wave = tid >> 5;
  const int wm   = wave >> 2;   // 0..1
  const int wn   = wave & 3;    // 0..3
  const int half = lane >> 4;   // 0/1
  const int l16  = lane & 15;
  const int n0 = blockIdx.x * 128;
  const int s0 = blockIdx.y * 128;

  const v8f zero8 = {0.f,0.f,0.f,0.f,0.f,0.f,0.f,0.f};
  v8f acc[4][2];
#pragma unroll
  for (int a = 0; a < 4; ++a)
#pragma unroll
    for (int b = 0; b < 2; ++b) acc[a][b] = zero8;

  const u32x4 z4 = {0u,0u,0u,0u};

  // stage one K-tile (BKCONV wide) into buffer bi
  auto stage = [&](int bi, int kb) {
    const int k0 = kb * BKCONV;
    if (tid < 128) {
      const int r = tid;
      const int s = s0 + r;
      const int h = s >> 7, w = s & 127;
      const int t  = k0 >> 10;            // conv tap 0..8 (constant over tile)
      const int cb = k0 & 1023;
      const int dy = t / 3 - 1, dx = t % 3 - 1;
      const int hh = h + dy, ww = w + dx;
      const bool inb = (hh >= 0) & (hh < HDIM) & (ww >= 0) & (ww < WDIM);
      const __bf16* src = xT + ((size_t)(hh * WDIM + ww) * C_INC + cb);
      __bf16* dst = &As[bi][r][0];
      if (inb) {
#ifdef HAVE_ASYNC_LDS
#pragma unroll
        for (int q = 0; q < 8; ++q) async_cp16(src + q * 8, dst + q * 8);
#else
#pragma unroll
        for (int q = 0; q < 8; ++q)
          ((u32x4*)dst)[q] = ((const u32x4*)src)[q];
#endif
      } else {
#pragma unroll
        for (int q = 0; q < 8; ++q) ((u32x4*)dst)[q] = z4;   // zero-pad halo
      }
    } else {
      const int n = tid - 128;
      const __bf16* src = Bt + (size_t)(n0 + n) * KDIM + k0;
      __bf16* dst = &Bs[bi][n][0];
#ifdef HAVE_ASYNC_LDS
#pragma unroll
      for (int q = 0; q < 8; ++q) async_cp16(src + q * 8, dst + q * 8);
#else
#pragma unroll
      for (int q = 0; q < 8; ++q)
        ((u32x4*)dst)[q] = ((const u32x4*)src)[q];
#endif
    }
  };

  stage(0, 0);

  for (int kb = 0; kb < KT2; ++kb) {
    const int bi = kb & 1;
    WAIT_ASYNC0();              // current buffer's async transfers complete
    __syncthreads();            // visible to every wave
    if (kb + 1 < KT2) stage(bi ^ 1, kb + 1);   // prefetch next tile

#pragma unroll
    for (int kk = 0; kk < 2; ++kk) {           // two 16x16x32 sub-steps
      const int kkb = kk * 32;
      UFrag fa[4], fb[2];
#pragma unroll
      for (int mf = 0; mf < 4; ++mf) {         // A 16x32 bf16 frag layout
        const int row = wm * 64 + mf * 16 + l16;
        const int e0  = kkb + half * 8;
        fa[mf].u[0] = *(const u32x4*)&As[bi][row][e0];
        fa[mf].u[1] = *(const u32x4*)&As[bi][row][e0 + 16];
      }
#pragma unroll
      for (int nf = 0; nf < 2; ++nf) {         // B 32x16 bf16 frag layout
        const int row = wn * 32 + nf * 16 + l16;
        const int e0  = kkb + half * 8;
        fb[nf].u[0] = *(const u32x4*)&Bs[bi][row][e0];
        fb[nf].u[1] = *(const u32x4*)&Bs[bi][row][e0 + 16];
      }
#pragma unroll
      for (int mf = 0; mf < 4; ++mf)
#pragma unroll
        for (int nf = 0; nf < 2; ++nf)
          acc[mf][nf] = __builtin_amdgcn_wmma_f32_16x16x32_bf16(
              false, fa[mf].v, false, fb[nf].v, (short)0, acc[mf][nf], false, false);
    }
    __syncthreads();            // all reads of buffer bi done before its reuse
  }

  // epilogue: bias + ReLU -> feat bf16 [s][c_mid]
#pragma unroll
  for (int mf = 0; mf < 4; ++mf)
#pragma unroll
    for (int nf = 0; nf < 2; ++nf) {
      const int ng = n0 + wn * 32 + nf * 16 + l16;
      const float bias = bconv[ng];
#pragma unroll
      for (int r = 0; r < 8; ++r) {
        const int sg = s0 + wm * 64 + mf * 16 + r + half * 8;
        float v = acc[mf][nf][r] + bias;
        v = fmaxf(v, 0.f);
        feat[(size_t)sg * C_MIDC + ng] = (__bf16)v;
      }
    }
}

// ---------------- head GEMM (score 18 + loc 36, padded N=64) ----------------
__global__ __launch_bounds__(256) void k_head_wmma(
    const __bf16* __restrict__ feat, const __bf16* __restrict__ Wh,
    const float* __restrict__ biasAll, float* __restrict__ out) {
  __shared__ __align__(16) __bf16 As[128][32];
  __shared__ __align__(16) __bf16 Bs[64][32];

  const int tid  = threadIdx.x;
  const int lane = tid & 31;
  const int wave = tid >> 5;
  const int wm   = wave >> 1;   // 0..3
  const int wn   = wave & 1;    // 0..1
  const int half = lane >> 4;
  const int l16  = lane & 15;
  const int s0 = blockIdx.x * 128;

  const v8f zero8 = {0.f,0.f,0.f,0.f,0.f,0.f,0.f,0.f};
  v8f acc[2][2];
#pragma unroll
  for (int a = 0; a < 2; ++a)
#pragma unroll
    for (int b = 0; b < 2; ++b) acc[a][b] = zero8;

  for (int kb = 0; kb < 16; ++kb) {
    const int k0 = kb * 32;
    if (tid < 128) {
      const u32x4* src = (const u32x4*)(feat + (size_t)(s0 + tid) * C_MIDC + k0);
      u32x4* dst = (u32x4*)&As[tid][0];
#pragma unroll
      for (int q = 0; q < 4; ++q) dst[q] = src[q];
    } else if (tid < 192) {
      const int n = tid - 128;
      const u32x4* src = (const u32x4*)(Wh + (size_t)n * 512 + k0);
      u32x4* dst = (u32x4*)&Bs[n][0];
#pragma unroll
      for (int q = 0; q < 4; ++q) dst[q] = src[q];
    }
    __syncthreads();

    UFrag fa[2], fb[2];
#pragma unroll
    for (int mf = 0; mf < 2; ++mf) {
      const int row = wm * 32 + mf * 16 + l16;
      const int e0  = half * 8;
      fa[mf].u[0] = *(const u32x4*)&As[row][e0];
      fa[mf].u[1] = *(const u32x4*)&As[row][16 + e0];
    }
#pragma unroll
    for (int nf = 0; nf < 2; ++nf) {
      const int row = wn * 32 + nf * 16 + l16;
      const int e0  = half * 8;
      fb[nf].u[0] = *(const u32x4*)&Bs[row][e0];
      fb[nf].u[1] = *(const u32x4*)&Bs[row][e0 + 16];
    }
#pragma unroll
    for (int mf = 0; mf < 2; ++mf)
#pragma unroll
      for (int nf = 0; nf < 2; ++nf)
        acc[mf][nf] = __builtin_amdgcn_wmma_f32_16x16x32_bf16(
            false, fa[mf].v, false, fb[nf].v, (short)0, acc[mf][nf], false, false);
    __syncthreads();
  }

#pragma unroll
  for (int mf = 0; mf < 2; ++mf)
#pragma unroll
    for (int nf = 0; nf < 2; ++nf) {
      const int nl = wn * 32 + nf * 16 + l16;   // 0..63
      const float bias = biasAll[nl];
#pragma unroll
      for (int r = 0; r < 8; ++r) {
        const int sg = s0 + wm * 32 + mf * 16 + r + half * 8;
        const float v = acc[mf][nf][r] + bias;
        if (nl < 18)       out[SCORE_OFF + (size_t)sg * 18 + nl] = v;
        else if (nl < 54)  out[LOC_OFF   + (size_t)sg * 36 + (nl - 18)] = v;
      }
    }
}

// ---------------- proposal: softmax fg, loc2bbox, clip, filter --------------
__global__ void k_boxes(const float* __restrict__ out, const float* __restrict__ anchors,
                        const int* __restrict__ img, const float* __restrict__ scale,
                        float* __restrict__ boxes, float* __restrict__ key,
                        int* __restrict__ sidx) {
  int i = blockIdx.x * blockDim.x + threadIdx.x;
  if (i >= NPAD) return;
  if (i >= N_ANCH) { key[i] = -3.0e38f; sidx[i] = i; return; }
  const int s = i / 9, a = i - s * 9;
  const float x0 = out[SCORE_OFF + (size_t)s * 18 + a * 2];
  const float x1 = out[SCORE_OFF + (size_t)s * 18 + a * 2 + 1];
  const float m  = fmaxf(x0, x1);
  const float e0 = __expf(x0 - m), e1 = __expf(x1 - m);
  const float fg = e1 / (e0 + e1);

  const float* an = anchors + (size_t)i * 4;
  const float ah = an[2] - an[0], aw = an[3] - an[1];
  const float acy = an[0] + 0.5f * ah, acx = an[1] + 0.5f * aw;
  const float* lc = out + LOC_OFF + (size_t)i * 4;
  const float cy = lc[0] * ah + acy;
  const float cx = lc[1] * aw + acx;
  const float bh = __expf(lc[2]) * ah;
  const float bw = __expf(lc[3]) * aw;
  const float imh = (float)img[0], imw = (float)img[1];
  float y0 = fminf(fmaxf(cy - 0.5f * bh, 0.f), imh);
  float x0b = fminf(fmaxf(cx - 0.5f * bw, 0.f), imw);
  float y1 = fminf(fmaxf(cy + 0.5f * bh, 0.f), imh);
  float x1b = fminf(fmaxf(cx + 0.5f * bw, 0.f), imw);
  f32x4 b; b.x = y0; b.y = x0b; b.z = y1; b.w = x1b;
  ((f32x4*)boxes)[i] = b;
  const float ms = 16.0f * scale[0];
  const bool valid = ((y1 - y0) >= ms) & ((x1b - x0b) >= ms);
  key[i] = valid ? fg : -1.0e9f;
  sidx[i] = i;
}

// ---------------- bitonic sort (descending key, ascending idx tiebreak) -----
__global__ void k_bitonic(float* __restrict__ key, int* __restrict__ sidx,
                          unsigned j, unsigned k) {
  unsigned i = blockIdx.x * blockDim.x + threadIdx.x;
  unsigned ixj = i ^ j;
  if (ixj <= i || i >= NPAD) return;
  float ka = key[i], kb2 = key[ixj];
  int ia = sidx[i], ib = sidx[ixj];
  const bool aWorse = (ka < kb2) || (ka == kb2 && ia > ib);
  const bool up = ((i & k) == 0);
  if (up == aWorse) {
    key[i] = kb2; key[ixj] = ka;
    sidx[i] = ib; sidx[ixj] = ia;
  }
}

__global__ void k_gather(const float* __restrict__ boxes, const int* __restrict__ sidx,
                         float* __restrict__ bsort) {
  int i = blockIdx.x * blockDim.x + threadIdx.x;
  if (i >= PRE_NMSN) return;
  ((f32x4*)bsort)[i] = ((const f32x4*)boxes)[sidx[i]];
}

// ---------------- NMS bitmask ----------------
__global__ void k_mask(const float* __restrict__ bsort, unsigned* __restrict__ mask) {
  int id = blockIdx.x * blockDim.x + threadIdx.x;
  if (id >= PRE_NMSN * MASKW) return;
  const int i = id / MASKW, jb = id - i * MASKW;
  const f32x4 bi = ((const f32x4*)bsort)[i];
  const float ai = (bi.z - bi.x) * (bi.w - bi.y);
  unsigned m = 0;
  const int j0 = jb * 32;
#pragma unroll 4
  for (int b = 0; b < 32; ++b) {
    const int j = j0 + b;
    if (j <= i || j >= PRE_NMSN) continue;
    const f32x4 bj = ((const f32x4*)bsort)[j];
    const float aj = (bj.z - bj.x) * (bj.w - bj.y);
    const float ty = fmaxf(bi.x, bj.x), tx = fmaxf(bi.y, bj.y);
    const float by = fminf(bi.z, bj.z), bx = fminf(bi.w, bj.w);
    const float ih = fmaxf(by - ty, 0.f), iw = fmaxf(bx - tx, 0.f);
    const float inter = ih * iw;
    const float iou = inter / (ai + aj - inter + 1e-9f);
    if (iou > NMS_TH) m |= (1u << b);
  }
  mask[id] = m;
}

// ---------------- greedy NMS scan (single block) + roi writeout -------------
__global__ __launch_bounds__(256) void k_nms_scan(const unsigned* __restrict__ mask,
                                                  const float* __restrict__ bsort,
                                                  float* __restrict__ out) {
  __shared__ unsigned removed[MASKW];
  __shared__ int s_keep;
  __shared__ int s_cnt;
  const int tid = threadIdx.x;
  if (tid < MASKW) removed[tid] = 0u;
  if (tid == 0) s_cnt = 0;
  __syncthreads();
  for (int i = 0; i < PRE_NMSN; ++i) {
    if (tid == 0) s_keep = !((removed[i >> 5] >> (i & 31)) & 1u);
    __syncthreads();
    if (s_keep) {
      if (tid < MASKW) removed[tid] |= mask[(size_t)i * MASKW + tid];
      if (tid == 0 && s_cnt < POST_NMSN) {
        const f32x4 b = ((const f32x4*)bsort)[i];
        float* r = out + ROI_OFF + (size_t)s_cnt * 4;
        r[0] = b.x; r[1] = b.y; r[2] = b.z; r[3] = b.w;
        s_cnt++;
      }
    }
    __syncthreads();
  }
  const int cnt = s_cnt;
  for (int q = tid; q < POST_NMSN * 4; q += 256)
    if (q >= cnt * 4) out[ROI_OFF + q] = 0.f;
}

// ---------------- launch ----------------
extern "C" void kernel_launch(void* const* d_in, const int* in_sizes, int n_in,
                              void* d_out, int out_size, void* d_ws, size_t ws_size,
                              hipStream_t stream) {
  (void)in_sizes; (void)n_in; (void)out_size; (void)ws_size;
  const float* x       = (const float*)d_in[0];
  const float* anchors = (const float*)d_in[1];
  const int*   img     = (const int*)d_in[2];
  const float* scale   = (const float*)d_in[3];
  const float* Wc      = (const float*)d_in[4];
  const float* bc      = (const float*)d_in[5];
  const float* Ws      = (const float*)d_in[6];
  const float* bs      = (const float*)d_in[7];
  const float* Wl      = (const float*)d_in[8];
  const float* bl      = (const float*)d_in[9];
  float* out = (float*)d_out;
  char*  ws  = (char*)d_ws;

  __bf16* xT   = (__bf16*)(ws + XT_OFF);
  __bf16* Bt   = (__bf16*)(ws + BT_OFF);
  __bf16* feat = (__bf16*)(ws + FEAT_OFF);
  __bf16* Wh   = (__bf16*)(ws + WH_OFF);
  float*  biasAll = (float*)(ws + BIAS_OFF);
  float*  boxes = (float*)(ws + BOX_OFF);
  float*  key   = (float*)(ws + KEY_OFF);
  int*    sidx  = (int*)(ws + IDX_OFF);
  float*  bsort = (float*)(ws + BS_OFF);
  unsigned* mask = (unsigned*)(ws + MSK_OFF);

  // prep
  k_prep_x<<<(S_TOT * C_INC + 255) / 256, 256, 0, stream>>>(x, xT);
  k_prep_bt<<<(C_MIDC * KDIM + 255) / 256, 256, 0, stream>>>(Wc, Bt);
  k_prep_head<<<(64 * 512 + 255) / 256, 256, 0, stream>>>(Ws, bs, Wl, bl, Wh, biasAll);

  // 3x3 conv (implicit GEMM, WMMA bf16, async double-buffered LDS)
  {
    dim3 grid(C_MIDC / 128, S_TOT / 128);   // (4, 128)
    k_conv_wmma<<<grid, 256, 0, stream>>>(xT, Bt, bc, feat);
  }

  // heads (WMMA bf16)
  k_head_wmma<<<S_TOT / 128, 256, 0, stream>>>(feat, Wh, biasAll, out);

  // proposal prep
  k_boxes<<<NPAD / 256, 256, 0, stream>>>(out, anchors, img, scale, boxes, key, sidx);

  // bitonic sort descending
  for (unsigned k = 2; k <= NPAD; k <<= 1)
    for (unsigned j = k >> 1; j > 0; j >>= 1)
      k_bitonic<<<NPAD / 256, 256, 0, stream>>>(key, sidx, j, k);

  k_gather<<<(PRE_NMSN + 255) / 256, 256, 0, stream>>>(boxes, sidx, bsort);
  k_mask<<<(PRE_NMSN * MASKW + 255) / 256, 256, 0, stream>>>(bsort, mask);
  k_nms_scan<<<1, 256, 0, stream>>>(mask, bsort, out);
}
